// DecoderOnlyAttention_38362647888634
// MI455X (gfx1250) — matrix operand order repaired and checked by reading, not compiled, not run on hardware
//
#include <hip/hip_runtime.h>
#include <hip/hip_bf16.h>

// ---------------------------------------------------------------------------
// Types
// ---------------------------------------------------------------------------
typedef unsigned short ushort_t;
typedef __attribute__((ext_vector_type(16))) __bf16   v16bf;
typedef __attribute__((ext_vector_type(8)))  float    v8f;
typedef __attribute__((ext_vector_type(8)))  ushort_t v8u16;
typedef __attribute__((ext_vector_type(16))) ushort_t v16u16;
typedef __attribute__((ext_vector_type(4)))  ushort_t v4u16;
typedef __attribute__((ext_vector_type(4)))  unsigned int u32x4;
typedef __attribute__((ext_vector_type(8)))  int      i32x8;
typedef __attribute__((ext_vector_type(4)))  int      i32x4;

#define B_   2
#define S_   2048
#define D_   2048
#define H_   16
#define DH_  128
#define TD_  6144   // 3*D

__device__ __forceinline__ ushort_t f2bf(float f) {
  unsigned u = __builtin_bit_cast(unsigned, f);
  u += 0x7fffu + ((u >> 16) & 1u);          // round-to-nearest-even
  return (ushort_t)(u >> 16);
}

__device__ __forceinline__ v16bf mk16(v8u16 lo, v8u16 hi) {
  v16u16 t;
#pragma unroll
  for (int i = 0; i < 8; ++i) { t[i] = lo[i]; t[i + 8] = hi[i]; }
  return __builtin_bit_cast(v16bf, t);
}

__device__ __forceinline__ v8f vzero8() {
  v8f z;
#pragma unroll
  for (int i = 0; i < 8; ++i) z[i] = 0.f;
  return z;
}

// ---------------------------------------------------------------------------
// Tiled GEMM:  C[M,N] = A[M,K] @ W[K,N] + bias   (bf16 WMMA, fp32 accumulate)
//   A: fp32 (convert+stage) or bf16 (TDM DMA stage), W/bias: fp32,
//   C: bf16 or fp32 (template).
//   Block tile 128x128x32, 256 threads = 8 waves, wave tile 64x32.
// ---------------------------------------------------------------------------
template <bool A_F32, bool OUT_BF16>
__global__ __launch_bounds__(256) void gemm_bias_wmma(
    const void* __restrict__ A_, const float* __restrict__ W,
    const float* __restrict__ bias, void* __restrict__ C_,
    int M, int N, int K) {
  __shared__ __align__(16) ushort_t As[128][40];   // [m][k]  (pad 8 -> 80B rows)
  __shared__ __align__(16) ushort_t Bst[128][40];  // [n][k]  transposed W tile

  const int tid  = threadIdx.x;
  const int w    = tid >> 5;
  const int lane = tid & 31;
  const int l16  = lane & 15;
  const int hi   = lane >> 4;
  const int wm   = w >> 2;                 // 0..1  (M waves)
  const int wn   = w & 3;                  // 0..3  (N waves)
  const int m0   = blockIdx.y * 128;
  const int n0   = blockIdx.x * 128;

  v8f acc[4][2];
#pragma unroll
  for (int mi = 0; mi < 4; ++mi)
#pragma unroll
    for (int ni = 0; ni < 2; ++ni) acc[mi][ni] = vzero8();

  const int KT = K >> 5;
  for (int kt = 0; kt < KT; ++kt) {
    // ---- stage A tile (128x32) into LDS as bf16 ----
    if constexpr (A_F32) {
      const float* A = (const float*)A_;
#pragma unroll
      for (int p = 0; p < 4; ++p) {
        int row = (tid >> 3) + p * 32;
        int c4  = (tid & 7) * 4;
        const float4 v =
            *(const float4*)&A[(size_t)(m0 + row) * K + kt * 32 + c4];
        v4u16 pk;
        pk[0] = f2bf(v.x); pk[1] = f2bf(v.y);
        pk[2] = f2bf(v.z); pk[3] = f2bf(v.w);
        *(v4u16*)&As[row][c4] = pk;
      }
    } else {
      // ---- Tensor Data Mover: DMA the bf16 A tile (128x32) into LDS with
      //      hardware row padding (16-DWORD interval, 4-DWORD pad -> 40
      //      ushorts per row, matching As[128][40]).  One wave issues the
      //      descriptor (TDM ignores EXEC; operands are wave-uniform),
      //      waits on TENSORcnt, then the workgroup barrier publishes LDS.
      if (w == 0) {
        const ushort_t* A = (const ushort_t*)A_;
        unsigned long long ga =
            (unsigned long long)(uintptr_t)(A + (size_t)m0 * K + kt * 32);
        unsigned ldsa = (unsigned)(uintptr_t)&As[0][0];  // LDS byte offset
        u32x4 g0;
        g0[0] = 1u;                                   // count=1 valid D#
        g0[1] = ldsa;                                 // lds_addr
        g0[2] = (unsigned)(ga & 0xffffffffu);         // global_addr lo
        g0[3] = (unsigned)((ga >> 32) & 0x1ffffffu)   // global_addr hi
                | (2u << 30);                         // type=2 ("image")
        i32x8 g1;
        g1[0] = (int)((1u << 16)      // data_size = 2 bytes
                    | (1u << 20)      // pad_enable
                    | (3u << 22)      // pad_interval: 16 DWORDs (64B rows)
                    | (3u << 25));    // pad_amount:   4 DWORDs (16B pad)
        g1[1] = (int)((((unsigned)K) & 0xffffu) << 16);      // dim0[15:0]
        g1[2] = (int)(((((unsigned)K) >> 16) & 0xffffu)      // dim0[31:16]
                    | ((((unsigned)M) & 0xffffu) << 16));    // dim1[15:0]
        g1[3] = (int)(((((unsigned)M) >> 16) & 0xffffu)      // dim1[31:16]
                    | (32u << 16));                          // tile_dim0=32
        g1[4] = 128;                                         // tile_dim1=128
        g1[5] = K;                                           // dim0_stride lo
        g1[6] = 0;
        g1[7] = 0;
        i32x4 gz = {0, 0, 0, 0};
#if __clang_major__ >= 23
        i32x8 gz8 = {0, 0, 0, 0, 0, 0, 0, 0};
        __builtin_amdgcn_tensor_load_to_lds(g0, g1, gz, gz, gz8, 0);
#else
        __builtin_amdgcn_tensor_load_to_lds(g0, g1, gz, gz, 0);
#endif
        __builtin_amdgcn_s_wait_tensorcnt(0);
      }
    }
    // ---- stage W tile (32x128) transposed: Bst[n][k] (fp32 -> bf16) ----
#pragma unroll
    for (int p = 0; p < 4; ++p) {
      int kr = (tid >> 5) + p * 8;
      int c4 = (tid & 31) * 4;
      const float4 v =
          *(const float4*)&W[(size_t)(kt * 32 + kr) * N + n0 + c4];
      Bst[c4 + 0][kr] = f2bf(v.x);
      Bst[c4 + 1][kr] = f2bf(v.y);
      Bst[c4 + 2][kr] = f2bf(v.z);
      Bst[c4 + 3][kr] = f2bf(v.w);
    }
    // prefetch next A tile into L2 while we compute
    if (kt + 1 < KT) {
      if constexpr (A_F32)
        __builtin_prefetch((const float*)A_ + (size_t)(m0 + (tid >> 3)) * K +
                               (kt + 1) * 32, 0, 1);
      else
        __builtin_prefetch((const ushort_t*)A_ + (size_t)(m0 + (tid >> 2)) * K +
                               (kt + 1) * 32, 0, 1);
    }
    __syncthreads();

    // ---- fragments + 8 WMMAs (bf16 16x16x32) ----
    v16bf afr[4];
#pragma unroll
    for (int mi = 0; mi < 4; ++mi) {
      int row = wm * 64 + mi * 16 + l16;
      v8u16 lo = *(const v8u16*)&As[row][hi * 8];        // K = hi*8 .. +7
      v8u16 hv = *(const v8u16*)&As[row][16 + hi * 8];   // K = 16+hi*8 .. +7
      afr[mi] = mk16(lo, hv);
    }
    v16bf bfr[2];
#pragma unroll
    for (int ni = 0; ni < 2; ++ni) {
      int col = wn * 32 + ni * 16 + l16;
      v8u16 lo = *(const v8u16*)&Bst[col][hi * 8];       // K = hi*8 .. +7
      v8u16 hv = *(const v8u16*)&Bst[col][16 + hi * 8];  // K = 16+hi*8 .. +7
      bfr[ni] = mk16(lo, hv);
    }
#pragma unroll
    for (int mi = 0; mi < 4; ++mi)
#pragma unroll
      for (int ni = 0; ni < 2; ++ni)
        acc[mi][ni] = __builtin_amdgcn_wmma_f32_16x16x32_bf16(
            false, afr[mi], false, bfr[ni], (short)0, acc[mi][ni], false, false);
    __syncthreads();
  }

  // ---- epilogue: bias + store ----
#pragma unroll
  for (int mi = 0; mi < 4; ++mi) {
#pragma unroll
    for (int ni = 0; ni < 2; ++ni) {
      int col  = n0 + wn * 32 + ni * 16 + l16;
      float bv = bias[col];
#pragma unroll
      for (int r = 0; r < 8; ++r) {
        int row   = m0 + wm * 64 + mi * 16 + hi * 8 + r;
        float val = acc[mi][ni][r] + bv;
        if constexpr (OUT_BF16)
          ((ushort_t*)C_)[(size_t)row * N + col] = f2bf(val);
        else
          ((float*)C_)[(size_t)row * N + col] = val;
      }
    }
  }
}

// ---------------------------------------------------------------------------
// Flash attention: one block = (b, h, 128 queries); 8 waves x 16 query rows.
// qkv is bf16 [B,S,3D]; scores never touch global memory.
// ---------------------------------------------------------------------------
__global__ __launch_bounds__(256) void attn_flash_wmma(
    const ushort_t* __restrict__ qkv, const float* __restrict__ mask,
    ushort_t* __restrict__ attn_out) {
  __shared__ __align__(16) ushort_t Vt[128][72];  // [dh][kv]  (transposed V)
  __shared__ __align__(16) ushort_t Ps[128][72];  // [q_local][kv] wave-private rows

  const int tid  = threadIdx.x;
  const int w    = tid >> 5;
  const int lane = tid & 31;
  const int l16  = lane & 15;
  const int hi   = lane >> 4;
  const int b    = blockIdx.y >> 4;
  const int h    = blockIdx.y & 15;
  const int q0   = blockIdx.x * 128;

  // ---- Q fragments for this wave's 16 rows: dh=128 -> 4 K-chunks of 32 ----
  v16bf qf[4];
  {
    const ushort_t* qrow =
        qkv + (size_t)(b * S_ + q0 + w * 16 + l16) * TD_ + h * DH_;
#pragma unroll
    for (int kc = 0; kc < 4; ++kc) {
      v8u16 lo = *(const v8u16*)(qrow + kc * 32 + hi * 8);
      v8u16 hv = *(const v8u16*)(qrow + kc * 32 + 16 + hi * 8);
      qf[kc] = mk16(lo, hv);
    }
  }

  float mrow[8], lrow[8];
#pragma unroll
  for (int r = 0; r < 8; ++r) { mrow[r] = -3.0e38f; lrow[r] = 0.f; }
  v8f o[8];
#pragma unroll
  for (int f = 0; f < 8; ++f) o[f] = vzero8();

  const float scale = 0.08838834764831845f;  // 1/sqrt(128)

  for (int j = 0; j < S_ / 64; ++j) {
    __syncthreads();  // previous P@V done reading Vt
    // ---- stage V block [64 kv][128 dh] transposed into Vt[dh][kv] ----
    {
      int kv = tid >> 2;
      int ch = tid & 3;
      const ushort_t* vrow =
          qkv + (size_t)(b * S_ + j * 64 + kv) * TD_ + 2 * D_ + h * DH_;
#pragma unroll
      for (int p = 0; p < 4; ++p) {
        int d0 = p * 32 + ch * 8;
        v8u16 v = *(const v8u16*)(vrow + d0);
#pragma unroll
        for (int i = 0; i < 8; ++i) Vt[d0 + i][kv] = v[i];
      }
    }
    __syncthreads();

    // ---- scores S = (Q @ K^T) * scale + mask bias (K streamed from global) ----
    v8f sc[4];
#pragma unroll
    for (int nt = 0; nt < 4; ++nt) {
      v8f a = vzero8();
      const ushort_t* krow =
          qkv + (size_t)(b * S_ + j * 64 + nt * 16 + l16) * TD_ + D_ + h * DH_;
#pragma unroll
      for (int kc = 0; kc < 4; ++kc) {
        v8u16 lo = *(const v8u16*)(krow + kc * 32 + hi * 8);
        v8u16 hv = *(const v8u16*)(krow + kc * 32 + 16 + hi * 8);
        v16bf kf = mk16(lo, hv);
        a = __builtin_amdgcn_wmma_f32_16x16x32_bf16(false, qf[kc], false, kf,
                                                    (short)0, a, false, false);
      }
      float mv    = mask[b * S_ + j * 64 + nt * 16 + l16];
      float mbias = (1.f - mv) * -10000.f;
#pragma unroll
      for (int r = 0; r < 8; ++r) sc[nt][r] = a[r] * scale + mbias;
    }

    // ---- online softmax (rows 0-7 live in lanes 0-15, rows 8-15 in 16-31) ----
#pragma unroll
    for (int r = 0; r < 8; ++r) {
      float mx = fmaxf(fmaxf(sc[0][r], sc[1][r]), fmaxf(sc[2][r], sc[3][r]));
      mx = fmaxf(mx, __shfl_xor(mx, 1, 32));
      mx = fmaxf(mx, __shfl_xor(mx, 2, 32));
      mx = fmaxf(mx, __shfl_xor(mx, 4, 32));
      mx = fmaxf(mx, __shfl_xor(mx, 8, 32));
      float mnew = fmaxf(mrow[r], mx);
      float corr = __expf(mrow[r] - mnew);
      float rsum = 0.f;
#pragma unroll
      for (int nt = 0; nt < 4; ++nt) {
        float p = __expf(sc[nt][r] - mnew);
        sc[nt][r] = p;
        rsum += p;
      }
      rsum += __shfl_xor(rsum, 1, 32);
      rsum += __shfl_xor(rsum, 2, 32);
      rsum += __shfl_xor(rsum, 4, 32);
      rsum += __shfl_xor(rsum, 8, 32);
      lrow[r] = lrow[r] * corr + rsum;
      mrow[r] = mnew;
#pragma unroll
      for (int f = 0; f < 8; ++f) o[f][r] = o[f][r] * corr;
#pragma unroll
      for (int nt = 0; nt < 4; ++nt)
        Ps[w * 16 + hi * 8 + r][nt * 16 + l16] = f2bf(sc[nt][r]);
    }

    // wave-local LDS RAW: DS is in-order per wave; wait to be explicit.
    asm volatile("s_wait_dscnt 0" ::: "memory");

    // ---- O += P @ V  (K-dim = 64 kv -> 2 chunks of 32) ----
#pragma unroll
    for (int kc = 0; kc < 2; ++kc) {
      v8u16 plo = *(const v8u16*)&Ps[w * 16 + l16][kc * 32 + hi * 8];
      v8u16 phi = *(const v8u16*)&Ps[w * 16 + l16][kc * 32 + 16 + hi * 8];
      v16bf pf  = mk16(plo, phi);
#pragma unroll
      for (int f = 0; f < 8; ++f) {
        v8u16 vlo = *(const v8u16*)&Vt[f * 16 + l16][kc * 32 + hi * 8];
        v8u16 vhi = *(const v8u16*)&Vt[f * 16 + l16][kc * 32 + 16 + hi * 8];
        v16bf vf  = mk16(vlo, vhi);
        o[f] = __builtin_amdgcn_wmma_f32_16x16x32_bf16(false, pf, false, vf,
                                                       (short)0, o[f], false,
                                                       false);
      }
    }
  }

  // ---- normalize, store bf16 [B,S,D] (heads re-interleaved) ----
#pragma unroll
  for (int r = 0; r < 8; ++r) {
    float inv = lrow[r] > 0.f ? 1.f / lrow[r] : 0.f;
    size_t rowbase =
        (size_t)(b * S_ + q0 + w * 16 + hi * 8 + r) * D_ + h * DH_;
#pragma unroll
    for (int f = 0; f < 8; ++f)
      attn_out[rowbase + f * 16 + l16] = f2bf(o[f][r] * inv);
  }
}

// ---------------------------------------------------------------------------
// Launch
// ---------------------------------------------------------------------------
extern "C" void kernel_launch(void* const* d_in, const int* in_sizes, int n_in,
                              void* d_out, int out_size, void* d_ws,
                              size_t ws_size, hipStream_t stream) {
  (void)in_sizes; (void)n_in; (void)out_size; (void)ws_size;
  const float* hidden = (const float*)d_in[0];
  const float* amask  = (const float*)d_in[1];
  const float* w_attn = (const float*)d_in[2];
  const float* b_attn = (const float*)d_in[3];
  const float* w_proj = (const float*)d_in[4];
  const float* b_proj = (const float*)d_in[5];
  float* out = (float*)d_out;

  // workspace: qkv bf16 [B*S, 3D] = 48MB, attn_out bf16 [B*S, D] = 16MB
  ushort_t* qkv  = (ushort_t*)d_ws;
  ushort_t* aout = qkv + (size_t)(B_ * S_) * TD_;

  // 1) QKV projection: [4096,2048] @ [2048,6144] + bias -> bf16
  gemm_bias_wmma<true, true><<<dim3(TD_ / 128, (B_ * S_) / 128), 256, 0,
                               stream>>>(hidden, w_attn, b_attn, qkv,
                                         B_ * S_, TD_, D_);
  // 2) flash attention per (b,h), 128-query blocks
  attn_flash_wmma<<<dim3(S_ / 128, B_ * H_), 256, 0, stream>>>(qkv, amask,
                                                               aout);
  // 3) output projection (A staged by TDM): [4096,2048] @ [2048,2048] + bias
  gemm_bias_wmma<false, false><<<dim3(D_ / 128, (B_ * S_) / 128), 256, 0,
                                 stream>>>(aout, w_proj, b_proj, out,
                                           B_ * S_, D_, D_);
}
